// MMGCNModel_86646670230227
// MI455X (gfx1250) — matrix operand order repaired and checked by reading, not compiled, not run on hardware
//
#include <hip/hip_runtime.h>
#include <hip/hip_bf16.h>
#include <math.h>

#define NUSR 20000
#define NITM 40000
#define NN   (NUSR + NITM)   // 60000, divisible by 16
#define NE   1000000
#define NB   8192

typedef __attribute__((ext_vector_type(2)))  float    v2f;
typedef __attribute__((ext_vector_type(8)))  float    v8f;
typedef __attribute__((ext_vector_type(16))) __bf16   v16bf;
typedef __attribute__((ext_vector_type(4)))  unsigned u32x4;
typedef __attribute__((ext_vector_type(8)))  unsigned u32x8;

// ---- Tensor Data Mover: async 2D f32 tile (rows x tile_w) global -> LDS ----
// D# per CDNA5 ISA. The tile is presented as its own
// tensor (tensor_dim == tile_dim) with the real row stride, so no OOB is possible.
// data_size code 2 == 4 bytes. workgroup_mask must be 0 (not in a cluster).
__device__ __forceinline__ void tdm_load_2d_f32(unsigned lds_byte_off, const float* gptr,
                                                unsigned tile_w, unsigned rows,
                                                unsigned row_stride_elems) {
  unsigned long long ga = (unsigned long long)(uintptr_t)gptr;
  unsigned long long q0 = 1ull                                    // count=1, user desc
                        | ((unsigned long long)lds_byte_off << 32);
  unsigned long long q1 = (ga & 0x01FFFFFFFFFFFFFFull)            // global_addr[56:0] at bits 120:64
                        | (2ull << 62);                           // type=2 ("image")
  u32x4 g0 = {(unsigned)q0, (unsigned)(q0 >> 32), (unsigned)q1, (unsigned)(q1 >> 32)};

  unsigned long long s0 = (unsigned long long)row_stride_elems;   // tensor_dim0_stride (48b)
  unsigned long long p0 = (2ull << 16)                            // data_size = 4B
                        | ((unsigned long long)(tile_w & 0xFFFFu) << 48);  // tensor_dim0[15:0]
  unsigned long long p1 = (unsigned long long)(tile_w >> 16)      // tensor_dim0[31:16]
                        | ((unsigned long long)rows << 16)        // tensor_dim1 (bits 111:80)
                        | ((unsigned long long)tile_w << 48);     // tile_dim0 (bits 127:112)
  unsigned long long p2 = (unsigned long long)rows                // tile_dim1 (bits 143:128)
                        | ((s0 & 0xFFFFFFFFull) << 32);           // stride0[31:0] (bits 191:160)
  unsigned long long p3 = (s0 >> 32);                             // stride0[47:32] (bits 207:192)
  u32x8 g1 = {(unsigned)p0, (unsigned)(p0 >> 32), (unsigned)p1, (unsigned)(p1 >> 32),
              (unsigned)p2, (unsigned)(p2 >> 32), (unsigned)p3, (unsigned)(p3 >> 32)};
  asm volatile("tensor_load_to_lds %0, %1" :: "s"(g0), "s"(g1) : "memory");
}

// ---------------- elementwise / utility kernels ----------------

__global__ void zero_f32(float* __restrict__ p, int n) {
  int t = blockIdx.x * blockDim.x + threadIdx.x;
  if (t < n) p[t] = 0.0f;
}

__global__ void copy_f32(const float* __restrict__ in, float* __restrict__ out, int n) {
  int t = blockIdx.x * blockDim.x + threadIdx.x;
  if (t < n) out[t] = in[t];
}

// one-shot f32 -> bf16 (hardware cvt; RNE per C semantics)
__global__ void cvt_bf16(const float* __restrict__ in, __bf16* __restrict__ out, int n) {
  int t = blockIdx.x * blockDim.x + threadIdx.x;
  if (t < n) out[t] = (__bf16)in[t];
}

__global__ void build_ego_k(const float* __restrict__ Gu, const float* __restrict__ Gi,
                            float* __restrict__ ego) {
  int t = blockIdx.x * blockDim.x + threadIdx.x;
  if (t >= NN * 64) return;
  ego[t] = (t < NUSR * 64) ? Gu[t] : Gi[t - NUSR * 64];
}

__global__ void deg_count(const int* __restrict__ dst, float* __restrict__ deg) {
  int t = blockIdx.x * blockDim.x + threadIdx.x;
  if (t < NE) atomicAdd(&deg[dst[t]], 1.0f);
}

__global__ void deg_invert(const float* __restrict__ deg, float* __restrict__ dinv) {
  int t = blockIdx.x * blockDim.x + threadIdx.x;
  if (t < NN) dinv[t] = 1.0f / fmaxf(deg[t], 1.0f);
}

// one wave per row of 128: l2-normalize
__global__ void l2norm_kernel(const float* __restrict__ x, float* __restrict__ xn, int nrows) {
  int row  = (blockIdx.x * blockDim.x + threadIdx.x) >> 5;
  int lane = threadIdx.x & 31;
  if (row >= nrows) return;
  const float4 v = *(const float4*)(x + (size_t)row * 128 + lane * 4);
  float s = v.x * v.x + v.y * v.y + v.z * v.z + v.w * v.w;
  #pragma unroll
  for (int o = 16; o > 0; o >>= 1) s += __shfl_xor(s, o, 32);
  float inv = 1.0f / fmaxf(sqrtf(s), 1e-12f);
  float4 o4 = {v.x * inv, v.y * inv, v.z * inv, v.w * inv};
  *(float4*)(xn + (size_t)row * 128 + lane * 4) = o4;
}

// edge-parallel gather + scatter-add: hsum[dst] += y[src]; D/4 lanes per edge
template <int D>
__global__ void edge_scatter(const int* __restrict__ src, const int* __restrict__ dst,
                             const float* __restrict__ y, float* __restrict__ hsum, int nE) {
  const int CH = D / 4;
  int t = blockIdx.x * blockDim.x + threadIdx.x;
  int e = t / CH, c = t % CH;
  if (e >= nE) return;
  int s = src[e], d = dst[e];
  const float4 v = *(const float4*)(y + (size_t)s * D + c * 4);
  float* o = hsum + (size_t)d * D + c * 4;
  atomicAdd(o + 0, v.x);
  atomicAdd(o + 1, v.y);
  atomicAdd(o + 2, v.z);
  atomicAdd(o + 3, v.w);
}

// one wave per batch element: dot over 64 features
__global__ void dot_kernel(const int* __restrict__ uidx, const int* __restrict__ iidx,
                           const float* __restrict__ xall, float* __restrict__ out, int nB) {
  int b    = (blockIdx.x * blockDim.x + threadIdx.x) >> 5;
  int lane = threadIdx.x & 31;
  if (b >= nB) return;
  const float* gu = xall + (size_t)uidx[b] * 64;
  const float* gi = xall + (size_t)(NUSR + iidx[b]) * 64;
  float s = gu[lane] * gi[lane] + gu[lane + 32] * gi[lane + 32];
  #pragma unroll
  for (int o = 16; o > 0; o >>= 1) s += __shfl_xor(s, o, 32);
  if (lane == 0) out[b] = s;
}

// ---------------- bf16 WMMA + TDM double buffering: big feature projection ----
// out[row, 0..127] = F(row,:) @ W(128,KD)^T + bias, W pre-converted to bf16.
// 8 waves/block cover a 16x128 C tile. The 16xKC f32 A tile is DMA'd into LDS by
// the Tensor Data Mover (wave 0 issues; TENSORcnt + workgroup barriers publish),
// double buffered so TDM of chunk k+1 overlaps WMMA of chunk k.
// A fragment: k = 16*hi + {0..15} sequential -> one contiguous 16-float run (ds_load_b128 x4).
// B fragment: k = 16*hi + {0..15} sequential -> one contiguous 32B bf16 vector load.
template <int KD>
__global__ __launch_bounds__(256) void proj_gemm_bf16(const float* __restrict__ F,
                                                      const __bf16* __restrict__ Wbf,
                                                      const float* __restrict__ bias,
                                                      float* __restrict__ out, int nrows) {
  constexpr int KC = 64;                       // K chunk (floats)
  constexpr int NCH = KD / KC;
  __shared__ float sA[2][16 * KC];             // sole LDS object -> base offset 0

  int wave = threadIdx.x >> 5, lane = threadIdx.x & 31;
  int rlo = lane & 15, hi = lane >> 4;
  int row0 = blockIdx.x * 16;
  int col0 = wave * 16;
  const __bf16* bcol = Wbf + (size_t)(col0 + rlo) * KD;  // B[k][n] = W[n][k]

  if (wave == 0)
    tdm_load_2d_f32(0u, F + (size_t)row0 * KD, KC, 16, KD);

  v8f c = {};
  for (int ch = 0; ch < NCH; ++ch) {
    if (wave == 0) __builtin_amdgcn_s_wait_tensorcnt(0);  // chunk ch landed in LDS
    __syncthreads();
    if (wave == 0 && ch + 1 < NCH)                        // overlap next DMA with compute
      tdm_load_2d_f32((unsigned)(((ch + 1) & 1) * 16 * KC * 4),
                      F + (size_t)row0 * KD + (size_t)(ch + 1) * KC, KC, 16, KD);

    const float* arow = &sA[ch & 1][rlo * KC];
    const int kb = ch * KC;
    #pragma unroll
    for (int kk = 0; kk < KC; kk += 32) {
      const float* ap = arow + kk + 16 * hi;
      float4 f0 = *(const float4*)(ap);
      float4 f1 = *(const float4*)(ap + 4);
      float4 f2 = *(const float4*)(ap + 8);
      float4 f3 = *(const float4*)(ap + 12);
      v16bf a;
      a[0]  = (__bf16)f0.x; a[1]  = (__bf16)f0.y; a[2]  = (__bf16)f0.z; a[3]  = (__bf16)f0.w;
      a[4]  = (__bf16)f1.x; a[5]  = (__bf16)f1.y; a[6]  = (__bf16)f1.z; a[7]  = (__bf16)f1.w;
      a[8]  = (__bf16)f2.x; a[9]  = (__bf16)f2.y; a[10] = (__bf16)f2.z; a[11] = (__bf16)f2.w;
      a[12] = (__bf16)f3.x; a[13] = (__bf16)f3.y; a[14] = (__bf16)f3.z; a[15] = (__bf16)f3.w;
      v16bf b = *(const v16bf*)(bcol + kb + kk + 16 * hi);   // 2x global_load_b128
      c = __builtin_amdgcn_wmma_f32_16x16x32_bf16(false, a, false, b, (short)0, c, false, false);
    }
    __syncthreads();   // all waves done reading buf before TDM reuses it
  }

  int cn = col0 + rlo;
  float bv = bias[cn];
  #pragma unroll
  for (int r = 0; r < 8; ++r) {
    int crow = row0 + r + 8 * hi;
    if (crow < nrows) out[(size_t)crow * 128 + cn] = c[r] + bv;
  }
}

// ---------------- f32 WMMA: fused per-node GEMMs ----------------
// out = oscale * ( (A1*rowscale) @ B1 + A2 @ B2 + bias + extra ) [+ out if accum]
// B element [k][n] read as B[k*bsk + n*bsn]  (so W^T from row-major W(N,K) is bsk=1,bsn=K)
template <int K1, int K2, int NCOL>
__global__ __launch_bounds__((NCOL / 16) * 32) void fused_gemm_f32(
    const float* __restrict__ A1, const float* __restrict__ rowscale,
    const float* __restrict__ B1, int b1sk, int b1sn,
    const float* __restrict__ A2,
    const float* __restrict__ B2, int b2sk, int b2sn,
    const float* __restrict__ bias, const float* __restrict__ extra,
    float* __restrict__ out, float oscale, int accum) {
  int wave = threadIdx.x >> 5, lane = threadIdx.x & 31;
  int rlo = lane & 15, hi = lane >> 4;
  int row0 = blockIdx.x * 16;
  int col0 = wave * 16;
  int ar = row0 + rlo;
  int cn = col0 + rlo;
  float rs = rowscale ? rowscale[ar] : 1.0f;
  v8f c = {};
  const float* a1 = A1 + (size_t)ar * K1;
  #pragma unroll 4
  for (int k0 = 0; k0 < K1; k0 += 4) {   // V_WMMA_F32_16X16X4_F32: K = 2*hi + e
    int k = k0 + hi * 2;
    v2f a, b;
    a[0] = a1[k] * rs;
    a[1] = a1[k + 1] * rs;
    b[0] = B1[(size_t)k * b1sk + (size_t)cn * b1sn];
    b[1] = B1[(size_t)(k + 1) * b1sk + (size_t)cn * b1sn];
    c = __builtin_amdgcn_wmma_f32_16x16x4_f32(false, a, false, b, (short)0, c, false, false);
  }
  if (K2 > 0) {
    const float* a2 = A2 + (size_t)ar * K2;
    #pragma unroll 4
    for (int k0 = 0; k0 < K2; k0 += 4) {
      int k = k0 + hi * 2;
      v2f a, b;
      a[0] = a2[k];
      a[1] = a2[k + 1];
      b[0] = B2[(size_t)k * b2sk + (size_t)cn * b2sn];
      b[1] = B2[(size_t)(k + 1) * b2sk + (size_t)cn * b2sn];
      c = __builtin_amdgcn_wmma_f32_16x16x4_f32(false, a, false, b, (short)0, c, false, false);
    }
  }
  float bv = bias ? bias[cn] : 0.0f;
  #pragma unroll
  for (int r = 0; r < 8; ++r) {
    int crow = row0 + r + 8 * hi;
    float v = c[r] + bv;
    if (extra) v += extra[(size_t)crow * NCOL + cn];
    v *= oscale;
    size_t oi = (size_t)crow * NCOL + cn;
    out[oi] = accum ? (out[oi] + v) : v;
  }
}

// ---------------- host orchestration ----------------

static inline int cdiv(long long a, int b) { return (int)((a + b - 1) / b); }

extern "C" void kernel_launch(void* const* d_in, const int* in_sizes, int n_in,
                              void* d_out, int out_size, void* d_ws, size_t ws_size,
                              hipStream_t stream) {
  (void)in_sizes; (void)n_in; (void)out_size; (void)ws_size;
  const int* edge = (const int*)d_in[0];
  const int* src = edge;
  const int* dst = edge + NE;
  const int* uidx = (const int*)d_in[1];
  const int* iidx = (const int*)d_in[2];
  const float* Gu = (const float*)d_in[3];
  const float* Gi = (const float*)d_in[4];
  const float* Gum[2]   = {(const float*)d_in[5],  (const float*)d_in[6]};
  const float* Feat[2]  = {(const float*)d_in[7],  (const float*)d_in[8]};
  const float* projw[2] = {(const float*)d_in[9],  (const float*)d_in[11]};
  const float* projb[2] = {(const float*)d_in[10], (const float*)d_in[12]};
  const float* Wp0[2] = {(const float*)d_in[13], (const float*)d_in[15]};
  const float* Wp1[2] = {(const float*)d_in[14], (const float*)d_in[16]};
  const float* l0w[2] = {(const float*)d_in[17], (const float*)d_in[21]};
  const float* l0b[2] = {(const float*)d_in[18], (const float*)d_in[22]};
  const float* l1w[2] = {(const float*)d_in[19], (const float*)d_in[23]};
  const float* l1b[2] = {(const float*)d_in[20], (const float*)d_in[24]};
  const float* g0w[2] = {(const float*)d_in[25], (const float*)d_in[29]};
  const float* g0b[2] = {(const float*)d_in[26], (const float*)d_in[30]};
  const float* g1w[2] = {(const float*)d_in[27], (const float*)d_in[31]};
  const float* g1b[2] = {(const float*)d_in[28], (const float*)d_in[32]};

  float* w = (float*)d_ws;
  float* deg  = w;  w += NN;
  float* dinv = w;  w += NN;
  float* ego  = w;  w += (size_t)NN * 64;
  float* x0   = w;  w += (size_t)NN * 128;
  float* xn   = w;  w += (size_t)NN * 128;
  float* y    = w;  w += (size_t)NN * 128;
  float* hsum = w;  w += (size_t)NN * 128;
  float* xhat = w;  w += (size_t)NN * 64;
  float* x1   = w;  w += (size_t)NN * 64;
  float* xall = w;  w += (size_t)NN * 64;
  __bf16* wbf = (__bf16*)w;  w += (128 * 2048 * 2) / 4;   // bf16 copy of proj weights

  zero_f32<<<cdiv(NN, 256), 256, 0, stream>>>(deg, NN);
  deg_count<<<cdiv(NE, 256), 256, 0, stream>>>(dst, deg);
  deg_invert<<<cdiv(NN, 256), 256, 0, stream>>>(deg, dinv);
  build_ego_k<<<cdiv((long long)NN * 64, 256), 256, 0, stream>>>(Gu, Gi, ego);

  for (int m = 0; m < 2; ++m) {
    const int KD = (m == 0) ? 2048 : 768;
    // x0 = concat(Gum, F @ projw^T + projb) ; weights -> bf16 once
    cvt_bf16<<<cdiv((long long)128 * KD, 256), 256, 0, stream>>>(projw[m], wbf, 128 * KD);
    copy_f32<<<cdiv((long long)NUSR * 128, 256), 256, 0, stream>>>(Gum[m], x0, NUSR * 128);
    if (m == 0)
      proj_gemm_bf16<2048><<<NITM / 16, 256, 0, stream>>>(Feat[m], wbf, projb[m],
                                                          x0 + (size_t)NUSR * 128, NITM);
    else
      proj_gemm_bf16<768><<<NITM / 16, 256, 0, stream>>>(Feat[m], wbf, projb[m],
                                                         x0 + (size_t)NUSR * 128, NITM);
    l2norm_kernel<<<cdiv((long long)NN * 32, 256), 256, 0, stream>>>(x0, xn, NN);

    // layer 0: y = xn @ Wp0 ; h = scatter(y) ; xhat = xn@l0w^T+l0b+ego ;
    //          x1 = (h*dinv)@g0w[:,:128]^T + xhat@g0w[:,128:]^T + g0b
    fused_gemm_f32<128, 0, 128><<<NN / 16, 256, 0, stream>>>(
        xn, nullptr, Wp0[m], 128, 1, nullptr, nullptr, 0, 0, nullptr, nullptr, y, 1.0f, 0);
    zero_f32<<<cdiv((long long)NN * 128, 256), 256, 0, stream>>>(hsum, NN * 128);
    edge_scatter<128><<<cdiv((long long)NE * 32, 256), 256, 0, stream>>>(src, dst, y, hsum, NE);
    fused_gemm_f32<128, 0, 64><<<NN / 16, 128, 0, stream>>>(
        xn, nullptr, l0w[m], 1, 128, nullptr, nullptr, 0, 0, l0b[m], ego, xhat, 1.0f, 0);
    fused_gemm_f32<128, 64, 64><<<NN / 16, 128, 0, stream>>>(
        hsum, dinv, g0w[m], 1, 192, xhat, g0w[m] + 128, 1, 192, g0b[m], nullptr, x1, 1.0f, 0);

    // layer 1 (dim 64)
    fused_gemm_f32<64, 0, 64><<<NN / 16, 128, 0, stream>>>(
        x1, nullptr, Wp1[m], 64, 1, nullptr, nullptr, 0, 0, nullptr, nullptr, y, 1.0f, 0);
    zero_f32<<<cdiv((long long)NN * 64, 256), 256, 0, stream>>>(hsum, NN * 64);
    edge_scatter<64><<<cdiv((long long)NE * 16, 256), 256, 0, stream>>>(src, dst, y, hsum, NE);
    fused_gemm_f32<64, 0, 64><<<NN / 16, 128, 0, stream>>>(
        x1, nullptr, l1w[m], 1, 64, nullptr, nullptr, 0, 0, l1b[m], ego, xhat, 1.0f, 0);
    // xall  (+)=  0.5 * x2
    fused_gemm_f32<64, 64, 64><<<NN / 16, 128, 0, stream>>>(
        hsum, dinv, g1w[m], 1, 128, xhat, g1w[m] + 64, 1, 128, g1b[m], nullptr, xall, 0.5f, m);
  }

  dot_kernel<<<cdiv((long long)NB * 32, 256), 256, 0, stream>>>(uidx, iidx, xall, (float*)d_out, NB);
}
